// WaveVitAttention_10144712753801
// MI455X (gfx1250) — matrix operand order repaired and checked by reading, not compiled, not run on hardware
//
#include <hip/hip_runtime.h>
#include <hip/hip_bf16.h>
#include <stdint.h>

typedef __bf16 bf16_t;
typedef __attribute__((ext_vector_type(16))) __bf16 v16bf;
typedef __attribute__((ext_vector_type(8)))  float  v8f;
typedef unsigned int u32x4 __attribute__((ext_vector_type(4)));
typedef int          i32x8 __attribute__((ext_vector_type(8)));
typedef int          i32x4 __attribute__((ext_vector_type(4)));

#if defined(__has_builtin)
#  if __has_builtin(__builtin_amdgcn_tensor_load_to_lds) && \
      __has_builtin(__builtin_amdgcn_s_wait_tensorcnt)
#    define USE_TDM 1
#  endif
#endif
#ifndef USE_TDM
#  define USE_TDM 0
#endif

// ---------------------------------------------------------------------------
// Fragment loaders following CDNA5 WMMA VGPR layouts (wave32).
// A (16x32 bf16): lane<16 -> row M=lane, K=0..7 & 16..23 ; lane>=16 -> row
// M=lane-16, K=8..15 & 24..31.  B (32x16 bf16): lane holds column N=lane%16,
// K as for A.  C/D: 8 VGPRs, lanes 0-15 rows M=r, 16-31 M=r+8.
// ---------------------------------------------------------------------------
__device__ inline v16bf load_a_frag(const bf16_t* tile, int ld, int lane) {
  union { v16bf v; uint32_t u[8]; } f;
  int m  = lane & 15;
  int kh = (lane >> 4) * 4;                       // dword offset within row
  const uint32_t* ar = reinterpret_cast<const uint32_t*>(tile + (size_t)m * ld);
#pragma unroll
  for (int i = 0; i < 4; ++i) { f.u[i] = ar[kh + i]; f.u[4 + i] = ar[8 + kh + i]; }
  return f.v;
}

__device__ inline v16bf load_b_frag(const bf16_t* tile, int ld, int lane) {
  v16bf f;
  int n  = lane & 15;
  int kh = (lane >> 4) * 8;
#pragma unroll
  for (int i = 0; i < 8; ++i) {
    f[i]     = tile[(size_t)(kh + i) * ld + n];
    f[8 + i] = tile[(size_t)(16 + kh + i) * ld + n];
  }
  return f;
}

// ---------------------------------------------------------------------------
// Elementwise helpers
// ---------------------------------------------------------------------------
// 8 floats -> 8 bf16 per thread (vectorized, n must be a multiple of 8)
__global__ void cvt_bf16x8_kernel(const float* __restrict__ in,
                                  bf16_t* __restrict__ out, int n8) {
  int i = blockIdx.x * blockDim.x + threadIdx.x;
  if (i >= n8) return;
  const float4* p = reinterpret_cast<const float4*>(in) + (size_t)i * 2;
  float4 a = p[0], b = p[1];
  bf16_t t[8] = {(bf16_t)a.x, (bf16_t)a.y, (bf16_t)a.z, (bf16_t)a.w,
                 (bf16_t)b.x, (bf16_t)b.y, (bf16_t)b.z, (bf16_t)b.w};
  *reinterpret_cast<uint4*>(out + (size_t)i * 8) =
      *reinterpret_cast<const uint4*>(t);
}

// w is [Nout][K] row-major f32 ; out is [K][Nout] bf16 (GEMM B operand)
__global__ void wtrans_kernel(const float* __restrict__ w,
                              bf16_t* __restrict__ out, int N, int K) {
  int i = blockIdx.x * blockDim.x + threadIdx.x;
  if (i >= N * K) return;
  int n = i / K, k = i - n * K;
  out[(size_t)k * N + n] = (bf16_t)w[i];
}

// Haar DWT: red_rm (row-major rows=(b,h*56+w), cols=64 bf16) ->
// coeffs (B,256,28,28) bf16, channel blocks [ll,lh,hl,hh].
__global__ void dwt_kernel(const bf16_t* __restrict__ red,
                           bf16_t* __restrict__ co) {
  int i = blockIdx.x * blockDim.x + threadIdx.x;
  const int TOT = 16 * 64 * 784;
  if (i >= TOT) return;
  int b = i / (64 * 784);
  int r = i - b * 64 * 784;
  int c = r / 784;
  int p = r - c * 784;
  int h = p / 28, w = p - h * 28;
  size_t base = (size_t)b * 3136;
  float a  = (float)red[(base + (2 * h    ) * 56 + 2 * w    ) * 64 + c];
  float bb = (float)red[(base + (2 * h    ) * 56 + 2 * w + 1) * 64 + c];
  float cc = (float)red[(base + (2 * h + 1) * 56 + 2 * w    ) * 64 + c];
  float dd = (float)red[(base + (2 * h + 1) * 56 + 2 * w + 1) * 64 + c];
  size_t ob = ((size_t)b * 256) * 784 + (size_t)c * 784 + p;
  co[ob            ] = (bf16_t)((a + bb + cc + dd) * 0.5f);
  co[ob +  64 * 784] = (bf16_t)((a - bb + cc - dd) * 0.5f);
  co[ob + 128 * 784] = (bf16_t)((a + bb - cc - dd) * 0.5f);
  co[ob + 192 * 784] = (bf16_t)((a - bb - cc + dd) * 0.5f);
}

// im2col for the 3x3 pad=1 conv on (B,256,28,28) NCHW bf16 input.
// A3 rows = b*784 + p (p = h*28+w), cols k = ci*9 + kh*3 + kw (K=2304).
__global__ void im2col3_kernel(const bf16_t* __restrict__ X,
                               bf16_t* __restrict__ A3) {
  int i = blockIdx.x * blockDim.x + threadIdx.x;
  const int TOT = 12544 * 2304;
  if (i >= TOT) return;
  int row = i / 2304;
  int k   = i - row * 2304;
  int b = row / 784;
  int p = row - b * 784;
  int h = p / 28, w = p - h * 28;
  int ci  = k / 9;
  int rem = k - ci * 9;
  int kh = rem / 3, kw = rem - kh * 3;
  int ih = h + kh - 1, iw = w + kw - 1;
  bf16_t v = (bf16_t)0.f;
  if ((unsigned)ih < 28u && (unsigned)iw < 28u)
    v = X[((size_t)(b * 256 + ci)) * 784 + ih * 28 + iw];
  A3[(size_t)row * 2304 + k] = v;
}

// Haar IDWT from conv output in ROW-MAJOR layout (rows=(b,p), cols=256 bf16)
// -> concat buffer cols [256,320)
__global__ void idwt_kernel(const bf16_t* __restrict__ co,
                            bf16_t* __restrict__ cat) {
  int i = blockIdx.x * blockDim.x + threadIdx.x;
  const int TOT = 16 * 64 * 784;
  if (i >= TOT) return;
  int b = i / (64 * 784);
  int r = i - b * 64 * 784;
  int c = r / 784;
  int p = r - c * 784;
  int h = p / 28, w = p - h * 28;
  size_t ib = ((size_t)b * 784 + p) * 256 + c;
  float ll = (float)co[ib];
  float lh = (float)co[ib + 64];
  float hl = (float)co[ib + 128];
  float hh = (float)co[ib + 192];
  float a  = (ll + lh + hl + hh) * 0.5f;
  float bb = (ll - lh + hl - hh) * 0.5f;
  float cc = (ll + lh - hl - hh) * 0.5f;
  float dd = (ll - lh - hl + hh) * 0.5f;
  size_t rowbase = (size_t)b * 3136;
  cat[(rowbase + (2 * h    ) * 56 + 2 * w    ) * 320 + 256 + c] = (bf16_t)a;
  cat[(rowbase + (2 * h    ) * 56 + 2 * w + 1) * 320 + 256 + c] = (bf16_t)bb;
  cat[(rowbase + (2 * h + 1) * 56 + 2 * w    ) * 320 + 256 + c] = (bf16_t)cc;
  cat[(rowbase + (2 * h + 1) * 56 + 2 * w + 1) * 320 + 256 + c] = (bf16_t)dd;
}

// im2col for the 2x2/stride-2 conv reading ROW-MAJOR conv output:
// A2 (3136 x 1024): k = ci*4 + dy*2 + dx
__global__ void im2col_kve_kernel(const bf16_t* __restrict__ co,
                                  bf16_t* __restrict__ a2) {
  int i = blockIdx.x * blockDim.x + threadIdx.x;
  const int TOT = 3136 * 1024;
  if (i >= TOT) return;
  int row = i >> 10;
  int k   = i & 1023;
  int b = row / 196;
  int p = row - b * 196;
  int y = p / 14, x = p - y * 14;
  int ci = k >> 2;
  int dy = (k >> 1) & 1, dx = k & 1;
  a2[i] = co[((size_t)b * 784 + (2 * y + dy) * 28 + (2 * x + dx)) * 256 + ci];
}

// Row-wise LayerNorm over 256 cols, 8 rows per block (one per wave)
__global__ __launch_bounds__(256) void ln_kernel(const float* __restrict__ X,
                                                 const float* __restrict__ g,
                                                 const float* __restrict__ bta,
                                                 bf16_t* __restrict__ Y, int rows) {
  int row  = blockIdx.x * 8 + (threadIdx.x >> 5);
  int lane = threadIdx.x & 31;
  if (row >= rows) return;
  const float* x = X + (size_t)row * 256;
  float v[8];
  float s = 0.f;
#pragma unroll
  for (int i = 0; i < 8; ++i) { v[i] = x[lane + 32 * i]; s += v[i]; }
#pragma unroll
  for (int d = 16; d; d >>= 1) s += __shfl_xor(s, d, 32);
  float mu = s * (1.0f / 256.0f);
  float q = 0.f;
#pragma unroll
  for (int i = 0; i < 8; ++i) { float t = v[i] - mu; q += t * t; }
#pragma unroll
  for (int d = 16; d; d >>= 1) q += __shfl_xor(q, d, 32);
  float inv = rsqrtf(q * (1.0f / 256.0f) + 1e-5f);
  bf16_t* y = Y + (size_t)row * 256;
#pragma unroll
  for (int i = 0; i < 8; ++i) {
    int c = lane + 32 * i;
    y[c] = (bf16_t)((v[i] - mu) * inv * g[c] + bta[c]);
  }
}

// ---------------------------------------------------------------------------
// Generic bf16 WMMA GEMM: C[M,N] = A[M,K] * B[K,N]  (A,B row-major bf16)
// REQUIRES M%64==0, N%64==0, K%32==0 (all call sites conform).
// Double-buffered LDS staging: next tile's global_load_b128 issues before the
// WMMAs, one barrier per K-step.
// mode 0: store bf16 ; 1: +bias, bf16 ; 2: +bias, f32 ; 3: BN+ReLU, bf16
// ---------------------------------------------------------------------------
__global__ __launch_bounds__(256) void wmma_gemm_kernel(
    const bf16_t* __restrict__ A, const bf16_t* __restrict__ Bm,
    void* __restrict__ Cout, int M, int N, int K,
    const float* __restrict__ bias,
    const float* __restrict__ bg, const float* __restrict__ bb,
    const float* __restrict__ bm, const float* __restrict__ bv, int mode) {
  __shared__ bf16_t As[2][64][32];
  __shared__ bf16_t Bs[2][32][64];
  int tid  = threadIdx.x;
  int lane = tid & 31;
  int wave = tid >> 5;
  int wm = wave >> 1;          // M strip 0..3
  int wn = wave & 1;           // N half 0..1
  int M0 = blockIdx.y * 64;
  int N0 = blockIdx.x * 64;

  int ra = tid >> 2, ca = (tid & 3) * 8;   // A-tile: one uint4 per thread
  int rb = tid >> 3, cb = (tid & 7) * 8;   // B-tile: one uint4 per thread
  const bf16_t* Aptr = A + (size_t)(M0 + ra) * K + ca;
  const bf16_t* Bptr = Bm + (size_t)rb * N + N0 + cb;

  // prologue: stage tile 0
  *reinterpret_cast<uint4*>(&As[0][ra][ca]) =
      *reinterpret_cast<const uint4*>(Aptr);
  *reinterpret_cast<uint4*>(&Bs[0][rb][cb]) =
      *reinterpret_cast<const uint4*>(Bptr);
  __syncthreads();

  v8f acc0 = {}, acc1 = {};
  int nk = K >> 5;
  for (int kt = 0; kt < nk; ++kt) {
    int cur = kt & 1;
    uint4 na = {}, nb = {};
    if (kt + 1 < nk) {
      na = *reinterpret_cast<const uint4*>(Aptr + (size_t)(kt + 1) * 32);
      nb = *reinterpret_cast<const uint4*>(Bptr + (size_t)(kt + 1) * 32 * N);
      __builtin_prefetch(Aptr + (size_t)(kt + 2) * 32, 0, 1);
      __builtin_prefetch(Bptr + (size_t)(kt + 2) * 32 * N, 0, 1);
    }
    v16bf af = load_a_frag(&As[cur][wm * 16][0], 32, lane);
    v16bf b0 = load_b_frag(&Bs[cur][0][wn * 32], 64, lane);
    v16bf b1 = load_b_frag(&Bs[cur][0][wn * 32 + 16], 64, lane);
    acc0 = __builtin_amdgcn_wmma_f32_16x16x32_bf16(false, af, false, b0,
                                                   (short)0, acc0, false, false);
    acc1 = __builtin_amdgcn_wmma_f32_16x16x32_bf16(false, af, false, b1,
                                                   (short)0, acc1, false, false);
    if (kt + 1 < nk) {
      *reinterpret_cast<uint4*>(&As[cur ^ 1][ra][ca]) = na;
      *reinterpret_cast<uint4*>(&Bs[cur ^ 1][rb][cb]) = nb;
      __syncthreads();
    }
  }

  int baseM = M0 + wm * 16 + (lane >> 4) * 8;
  int n0    = N0 + wn * 32 + (lane & 15);
#pragma unroll
  for (int f = 0; f < 2; ++f) {
    v8f acc = f ? acc1 : acc0;
    int gn = n0 + f * 16;
    float cb2 = 0.f, sc = 1.f, sh = 0.f;
    if (mode == 1 || mode == 2) cb2 = bias[gn];
    if (mode == 3) {
      float inv = bg[gn] * rsqrtf(bv[gn] + 1e-5f);
      sc = inv; sh = bb[gn] - bm[gn] * inv;
    }
#pragma unroll
    for (int r = 0; r < 8; ++r) {
      int gm = baseM + r;
      float val = acc[r];
      if (mode == 1 || mode == 2) val += cb2;
      if (mode == 3) val = fmaxf(val * sc + sh, 0.f);
      if (mode == 2) ((float*)Cout)[(size_t)gm * N + gn] = val;
      else           ((bf16_t*)Cout)[(size_t)gm * N + gn] = (bf16_t)val;
    }
  }
}

// ---------------------------------------------------------------------------
// Attention: per-(b, head, 64-query tile) block, 128 threads = 4 waves.
// Q: row-major (B*N,256) bf16 ; KV: row-major (B*196,512) bf16 (key|value).
// V tile is fetched by the Tensor Data Mover (2D D# descriptor) when the
// builtin is available.  Writes attended into concat buffer cols [0,256).
// ---------------------------------------------------------------------------
__global__ __launch_bounds__(128) void attn_kernel(
    const bf16_t* __restrict__ Q, const bf16_t* __restrict__ KV,
    bf16_t* __restrict__ Cat) {
  const int N = 3136, MK = 196, MP = 224;
  int b  = blockIdx.z;
  int h  = blockIdx.y;
  int q0 = blockIdx.x * 64;
  __shared__ union ShU {
    struct { bf16_t Q[64][32]; bf16_t K[32][MP]; } qk;
    bf16_t V[MP][32];
  } U;
  __shared__ bf16_t Ps[64][MP];
  int tid = threadIdx.x, lane = tid & 31, wave = tid >> 5;  // 4 waves

  // Q: 64 rows x 4 uint4 chunks = 256 vector loads
  for (int u = tid; u < 64 * 4; u += 128) {
    int r = u >> 2, c8 = (u & 3) * 8;
    *reinterpret_cast<uint4*>(&U.qk.Q[r][c8]) = *reinterpret_cast<const uint4*>(
        Q + ((size_t)(b * N + q0 + r)) * 256 + h * 32 + c8);
  }
  // K^T: vector-load 8 d's for one m, scatter-transpose into LDS
  for (int u = tid; u < MP * 4; u += 128) {
    int m = u >> 2, d8 = (u & 3) * 8;
    bf16_t tmp[8];
    if (m < MK) {
      *reinterpret_cast<uint4*>(tmp) = *reinterpret_cast<const uint4*>(
          KV + ((size_t)(b * MK + m)) * 512 + h * 32 + d8);
    } else {
#pragma unroll
      for (int j = 0; j < 8; ++j) tmp[j] = (bf16_t)0.f;
    }
#pragma unroll
    for (int j = 0; j < 8; ++j) U.qk.K[d8 + j][m] = tmp[j];
  }
  __syncthreads();

  // QK^T: wave owns 16-row strip; 14 column tiles, single K step (hd=32)
  int strip = wave;
  v16bf af = load_a_frag(&U.qk.Q[strip * 16][0], 32, lane);
  v8f acc[14];
#pragma unroll
  for (int t = 0; t < 14; ++t) {
    v16bf bt = load_b_frag(&U.qk.K[0][t * 16], MP, lane);
    v8f z = {};
    acc[t] = __builtin_amdgcn_wmma_f32_16x16x32_bf16(false, af, false, bt,
                                                     (short)0, z, false, false);
  }

  // Softmax in registers; rows live per (r, lane>>4); cols over (t, lane&15)
  const float scale = 0.17677669529663687f;  // 32^-0.5
  int ln = lane & 15;
#pragma unroll
  for (int r = 0; r < 8; ++r) {
    float e[14];
    float mx = -1e30f;
#pragma unroll
    for (int t = 0; t < 14; ++t) {
      int col = t * 16 + ln;
      float s = (col < MK) ? acc[t][r] * scale : -1e30f;
      e[t] = s;
      mx = fmaxf(mx, s);
    }
#pragma unroll
    for (int d = 8; d; d >>= 1) mx = fmaxf(mx, __shfl_xor(mx, d, 32));
    float sum = 0.f;
#pragma unroll
    for (int t = 0; t < 14; ++t) { e[t] = __expf(e[t] - mx); sum += e[t]; }
#pragma unroll
    for (int d = 8; d; d >>= 1) sum += __shfl_xor(sum, d, 32);
    float inv = 1.f / sum;
    int row = strip * 16 + r + 8 * (lane >> 4);
#pragma unroll
    for (int t = 0; t < 14; ++t) Ps[row][t * 16 + ln] = (bf16_t)(e[t] * inv);
  }
  __syncthreads();  // all reads of U.qk done; V region free

#if USE_TDM
  // TDM: 2D tile, 196 rows x 32 bf16, row stride 512 elements, contiguous in
  // LDS at 64 B/row (matches U.V layout).  Issued by wave 0 only (wave-level,
  // EXEC-ignored); barrier publishes to the workgroup.
  if (wave == 0) {
    uint64_t ga = (uint64_t)(uintptr_t)(KV + ((size_t)(b * MK)) * 512 +
                                        256 + h * 32);
    uint32_t lva = (uint32_t)(uintptr_t)(&U.V[0][0]);
    // D# group 0: count=1 | lds_addr | global_addr[56:0] | type=2
    u32x4 g0 = {1u, lva, (uint32_t)ga,
                (uint32_t)(((ga >> 32) & 0x01FFFFFFu) | 0x80000000u)};
    // D# group 1: data_size=2B ; tensor_dim0=32 ; tensor_dim1=196 ;
    // tile_dim0=32 ; tile_dim1=196 ; tensor_dim0_stride=512
    i32x8 g1 = {(int)(1u << 16),          // workgroup_mask=0, data_size=1(2B)
                (int)(32u << 16),         // tensor_dim0[15:0] in [31:16]
                (int)(196u << 16),        // tensor_dim0[31:16]=0, tensor_dim1 lo
                (int)(32u << 16),         // tensor_dim1 hi=0, tile_dim0=32
                (int)196,                 // tile_dim1=196, tile_dim2=0
                (int)512,                 // tensor_dim0_stride[31:0]
                0, 0};                    // stride hi / tensor_dim1_stride = 0
    i32x4 gz = {0, 0, 0, 0};
#if __clang_major__ >= 23
    i32x8 gz8 = {0, 0, 0, 0, 0, 0, 0, 0};
    __builtin_amdgcn_tensor_load_to_lds(g0, g1, gz, gz, gz8, 0);
#else
    __builtin_amdgcn_tensor_load_to_lds(g0, g1, gz, gz, 0);
#endif
  }
  // zero-fill pad rows [196,224) (disjoint from the TDM target region)
  for (int u = tid; u < 28 * 4; u += 128) {
    int m = 196 + (u >> 2), d8 = (u & 3) * 8;
    *reinterpret_cast<uint4*>(&U.V[m][d8]) = uint4{0, 0, 0, 0};
  }
  if (wave == 0) __builtin_amdgcn_s_wait_tensorcnt(0);
  __syncthreads();
#else
  for (int u = tid; u < MP * 4; u += 128) {
    int m = u >> 2, d8 = (u & 3) * 8;
    uint4 val = uint4{0, 0, 0, 0};
    if (m < MK)
      val = *reinterpret_cast<const uint4*>(
          KV + ((size_t)(b * MK + m)) * 512 + 256 + h * 32 + d8);
    *reinterpret_cast<uint4*>(&U.V[m][d8]) = val;
  }
  __syncthreads();
#endif

  // PV: strip x 2 d-tiles, 7 K steps of 32
  v8f o0 = {}, o1 = {};
#pragma unroll
  for (int kc = 0; kc < 7; ++kc) {
    v16bf pa = load_a_frag(&Ps[strip * 16][kc * 32], MP, lane);
    v16bf v0 = load_b_frag(&U.V[kc * 32][0], 32, lane);
    v16bf v1 = load_b_frag(&U.V[kc * 32][16], 32, lane);
    o0 = __builtin_amdgcn_wmma_f32_16x16x32_bf16(false, pa, false, v0, (short)0,
                                                 o0, false, false);
    o1 = __builtin_amdgcn_wmma_f32_16x16x32_bf16(false, pa, false, v1, (short)0,
                                                 o1, false, false);
  }
  int rb = q0 + strip * 16 + (lane >> 4) * 8;
  int c0 = h * 32 + ln;
#pragma unroll
  for (int r = 0; r < 8; ++r) {
    size_t base = ((size_t)(b * N + rb + r)) * 320;
    Cat[base + c0]      = (bf16_t)o0[r];
    Cat[base + c0 + 16] = (bf16_t)o1[r];
  }
}

// ---------------------------------------------------------------------------
extern "C" void kernel_launch(void* const* d_in, const int* in_sizes, int n_in,
                              void* d_out, int out_size, void* d_ws,
                              size_t ws_size, hipStream_t stream) {
  (void)in_sizes; (void)n_in; (void)out_size; (void)ws_size;
  const float* x        = (const float*)d_in[0];
  const float* reduce_w = (const float*)d_in[3];
  const float* bn1_g = (const float*)d_in[4];
  const float* bn1_b = (const float*)d_in[5];
  const float* bn1_m = (const float*)d_in[6];
  const float* bn1_v = (const float*)d_in[7];
  const float* filt_w = (const float*)d_in[8];
  const float* bn2_g = (const float*)d_in[9];
  const float* bn2_b = (const float*)d_in[10];
  const float* bn2_m = (const float*)d_in[11];
  const float* bn2_v = (const float*)d_in[12];
  const float* kve_w = (const float*)d_in[13];
  const float* kve_b = (const float*)d_in[14];
  const float* q_w   = (const float*)d_in[15];
  const float* q_b   = (const float*)d_in[16];
  const float* kvn_g = (const float*)d_in[17];
  const float* kvn_b = (const float*)d_in[18];
  const float* kv_w  = (const float*)d_in[19];
  const float* kv_b  = (const float*)d_in[20];
  const float* proj_w = (const float*)d_in[21];
  const float* proj_b = (const float*)d_in[22];

  const size_t BN = 50176;  // 16*3136

  char* ws = (char*)d_ws;
  size_t off = 0;
  auto alloc = [&](size_t bytes) -> void* {
    void* p = ws + off;
    off = (off + bytes + 255) & ~(size_t)255;
    return p;
  };
  bf16_t* xb     = (bf16_t*)alloc(BN * 256 * 2);
  bf16_t* qrm    = (bf16_t*)alloc(BN * 256 * 2);
  bf16_t* redrm  = (bf16_t*)alloc(BN * 64 * 2);
  bf16_t* co1    = (bf16_t*)alloc((size_t)16 * 256 * 784 * 2);   // NCHW
  bf16_t* a3     = (bf16_t*)alloc((size_t)12544 * 2304 * 2);     // conv3 im2col
  bf16_t* co2    = (bf16_t*)alloc((size_t)12544 * 256 * 2);      // row-major
  bf16_t* cat    = (bf16_t*)alloc(BN * 320 * 2);
  bf16_t* a2     = (bf16_t*)alloc((size_t)3136 * 1024 * 2);
  float*  kvsrc  = (float*) alloc((size_t)3136 * 256 * 4);
  bf16_t* lnb    = (bf16_t*)alloc((size_t)3136 * 256 * 2);
  bf16_t* kvbuf  = (bf16_t*)alloc((size_t)3136 * 512 * 2);
  bf16_t* qB     = (bf16_t*)alloc((size_t)256 * 256 * 2);
  bf16_t* redB   = (bf16_t*)alloc((size_t)256 * 64 * 2);
  bf16_t* filtB  = (bf16_t*)alloc((size_t)2304 * 256 * 2);
  bf16_t* kveB   = (bf16_t*)alloc((size_t)1024 * 256 * 2);
  bf16_t* kvB    = (bf16_t*)alloc((size_t)256 * 512 * 2);
  bf16_t* projB  = (bf16_t*)alloc((size_t)320 * 256 * 2);

  const int T = 256;
  // conversions / weight transposes
  cvt_bf16x8_kernel<<<(int)((BN * 256 / 8 + T - 1) / T), T, 0, stream>>>(
      x, xb, (int)(BN * 256 / 8));
  wtrans_kernel<<<(256 * 256 + T - 1) / T, T, 0, stream>>>(q_w, qB, 256, 256);
  wtrans_kernel<<<(64 * 256 + T - 1) / T, T, 0, stream>>>(reduce_w, redB, 64, 256);
  wtrans_kernel<<<(256 * 2304 + T - 1) / T, T, 0, stream>>>(filt_w, filtB, 256, 2304);
  wtrans_kernel<<<(256 * 1024 + T - 1) / T, T, 0, stream>>>(kve_w, kveB, 256, 1024);
  wtrans_kernel<<<(512 * 256 + T - 1) / T, T, 0, stream>>>(kv_w, kvB, 512, 256);
  wtrans_kernel<<<(256 * 320 + T - 1) / T, T, 0, stream>>>(proj_w, projB, 256, 320);

  // q projection: (50176x256)@(256x256)+bias -> bf16
  wmma_gemm_kernel<<<dim3(4, 784), T, 0, stream>>>(
      xb, qB, qrm, (int)BN, 256, 256, q_b, nullptr, nullptr, nullptr, nullptr, 1);
  // reduce 1x1 conv + BN + ReLU: (50176x256)@(256x64) -> bf16 row-major
  wmma_gemm_kernel<<<dim3(1, 784), T, 0, stream>>>(
      xb, redB, redrm, (int)BN, 64, 256, nullptr, bn1_g, bn1_b, bn1_m, bn1_v, 3);
  // Haar DWT -> NCHW bf16
  dwt_kernel<<<(16 * 64 * 784 + T - 1) / T, T, 0, stream>>>(redrm, co1);
  // 3x3 conv: explicit im2col, then GEMM(+BN+ReLU) -> row-major bf16
  im2col3_kernel<<<(12544 * 2304 + T - 1) / T, T, 0, stream>>>(co1, a3);
  wmma_gemm_kernel<<<dim3(4, 196), T, 0, stream>>>(
      a3, filtB, co2, 12544, 256, 2304, nullptr, bn2_g, bn2_b, bn2_m, bn2_v, 3);
  // IDWT -> concat cols [256,320)
  idwt_kernel<<<(16 * 64 * 784 + T - 1) / T, T, 0, stream>>>(co2, cat);
  // kve 2x2 stride-2 conv: im2col then GEMM (+bias, f32 out)
  im2col_kve_kernel<<<(3136 * 1024 + T - 1) / T, T, 0, stream>>>(co2, a2);
  wmma_gemm_kernel<<<dim3(4, 49), T, 0, stream>>>(
      a2, kveB, kvsrc, 3136, 256, 1024, kve_b, nullptr, nullptr, nullptr, nullptr, 2);
  // LayerNorm -> bf16
  ln_kernel<<<392, T, 0, stream>>>(kvsrc, kvn_g, kvn_b, lnb, 3136);
  // kv projection: (3136x256)@(256x512)+bias -> bf16 (key|value)
  wmma_gemm_kernel<<<dim3(8, 49), T, 0, stream>>>(
      lnb, kvB, kvbuf, 3136, 512, 256, kv_b, nullptr, nullptr, nullptr, nullptr, 1);
  // attention -> concat cols [0,256)
  attn_kernel<<<dim3(49, 8, 16), 128, 0, stream>>>(qrm, kvbuf, cat);
  // output projection: (50176x320)@(320x256)+bias -> f32 d_out
  wmma_gemm_kernel<<<dim3(4, 784), T, 0, stream>>>(
      cat, projB, d_out, (int)BN, 256, 320, proj_b, nullptr, nullptr, nullptr,
      nullptr, 2);
}
